// RandomWalkEmbed_86517821213148
// MI455X (gfx1250) — matrix operand order repaired and checked by reading, not compiled, not run on hardware
//
#include <hip/hip_runtime.h>
#include <hip/hip_bf16.h>
#include <math.h>

// ---------------------------------------------------------------------------
// RandomWalkEmbed p2g forward for MI455X (gfx1250), wave32 + WMMA bf16.
//
// probe [8,2048,144], gallery [32,2048,144], out [8,32,2].
// Dominant cost: 256 pair-GEMMs 144x144x2048 (~21.7 GFLOP) on
// v_wmma_f32_16x16x32_bf16. Working set ~24 MB -> L2-resident (192 MB),
// so the GEMM loop reads fragments straight from global (WGP$/L2 hits),
// no LDS staging, no barriers in the K-loop.
// Key simplification: warped_mean needs only softmax column sums:
//   s[b] = sum_a softmax_b(corr[a,:])[b];  warped_mean[c] = (1/HW) p[c,:]·s
// ---------------------------------------------------------------------------

typedef __attribute__((ext_vector_type(16))) __bf16 v16bf;
typedef __attribute__((ext_vector_type(8)))  float  v8f;

#define NP   8
#define NG   32
#define CCH  2048
#define HWSZ 144
#define NT   9            // 144/16 tiles per spatial dim
#define KSTEP 32
#define PAIR_THREADS 288  // 9 waves: wave w owns corr tile-row w
#define TB 16             // transpose tile: spatial
#define TC 128            // transpose tile: channels

__device__ __forceinline__ __bf16 f2bf(float f) {
    unsigned int u = __builtin_bit_cast(unsigned int, f);
    unsigned int r = (u + 0x7fffu + ((u >> 16) & 1u)) >> 16;   // RNE
    return __builtin_bit_cast(__bf16, (unsigned short)r);
}

// --- Stage 1a: per-pixel inverse L2 norm along C ---------------------------
__global__ __launch_bounds__(160)
void norminv_kernel(const float* __restrict__ x, float* __restrict__ ninv) {
    const int n = blockIdx.x;
    const int b = threadIdx.x;
    if (b >= HWSZ) return;
    const float* base = x + (size_t)n * CCH * HWSZ + b;
    float acc = 0.f;
    #pragma unroll 8
    for (int c = 0; c < CCH; ++c) {
        float v = base[(size_t)c * HWSZ];
        acc = fmaf(v, v, acc);
    }
    ninv[n * HWSZ + b] = rsqrtf(acc);
}

// --- Stage 1b: LDS-tiled transpose + normalize + bf16 convert --------------
// in:  x    [n][C][HW]  f32
// out: xt   [n][HW][C]  bf16   (rows = spatial, contiguous channels ->
//                               WMMA fragments become contiguous 16B loads)
__global__ __launch_bounds__(256)
void transpose_norm_kernel(const float* __restrict__ x,
                           const float* __restrict__ ninv,
                           __bf16* __restrict__ xt) {
    __shared__ float tile[TB * (TC + 1)];
    const int n  = blockIdx.x;
    const int b0 = blockIdx.y * TB;
    const int c0 = blockIdx.z * TC;
    const int tid = threadIdx.x;
    for (int idx = tid; idx < TB * TC; idx += 256) {       // coalesced reads
        const int bl = idx & (TB - 1);
        const int cl = idx >> 4;
        tile[bl * (TC + 1) + cl] = x[((size_t)n * CCH + c0 + cl) * HWSZ + b0 + bl];
    }
    __syncthreads();
    for (int idx = tid; idx < TB * TC; idx += 256) {       // coalesced writes
        const int cl = idx & (TC - 1);
        const int bl = idx >> 7;
        const float s = ninv[n * HWSZ + b0 + bl];
        xt[((size_t)n * HWSZ + b0 + bl) * CCH + c0 + cl] =
            f2bf(tile[bl * (TC + 1) + cl] * s);
    }
}

// --- Stage 1c: gallery spatial mean: gm[j,c] = mean_b g[j,c,b] -------------
__global__ __launch_bounds__(256)
void gmean_kernel(const float* __restrict__ g, float* __restrict__ gm) {
    const int idx = blockIdx.x * blockDim.x + threadIdx.x;   // j*CCH + c
    if (idx >= NG * CCH) return;
    const float* row = g + (size_t)idx * HWSZ;
    float s = 0.f;
    #pragma unroll 8
    for (int b = 0; b < HWSZ; ++b) s += row[b];
    gm[idx] = s * (1.0f / HWSZ);
}

// --- Stage 2: per-pair corr GEMM + softmax-colsum + fused epilogue ---------
__global__ __launch_bounds__(PAIR_THREADS)
void rw_pair_kernel(const float*  __restrict__ probe,
                    const __bf16* __restrict__ PNt,   // [i][b][c] bf16, normalized
                    const __bf16* __restrict__ GNt,   // [j][a][c] bf16, normalized
                    const float*  __restrict__ GM,
                    const float*  __restrict__ bn_gamma,
                    const float*  __restrict__ bn_beta,
                    const float*  __restrict__ bn_mean,
                    const float*  __restrict__ bn_var,
                    const float*  __restrict__ cls_w,
                    const float*  __restrict__ cls_b,
                    float*        __restrict__ out) {
    __shared__ float ldsSp[NT][HWSZ];            // per-wave s partials (deterministic)
    __shared__ float ldsS[HWSZ];                 // s[b] = sum_a att[a,b]
    __shared__ float red0[PAIR_THREADS];
    __shared__ float red1[PAIR_THREADS];

    const int pair = blockIdx.x;
    const int i = pair / NG, j = pair % NG;
    const int tid  = threadIdx.x;
    const int wave = tid >> 5;
    const int lane = tid & 31;
    const int half = lane >> 4;                  // 0: lanes 0-15, 1: lanes 16-31
    const int l16  = lane & 15;

    v8f acc[NT];
    #pragma unroll
    for (int t = 0; t < NT; ++t)
        #pragma unroll
        for (int e = 0; e < 8; ++e) acc[t][e] = 0.f;

    // A fragment (16x32 bf16) ISA layout: lane half 0 -> k {0..7, 16..23},
    // half 1 -> k {8..15, 24..31}: two contiguous 16B loads at +0 / +32B.
    const __bf16* Arow = GNt + ((size_t)j * HWSZ + wave * 16 + l16) * CCH + half * 8;
    // B fragment (32x16 bf16): same k order as A: half 0 -> k {0..7, 16..23},
    // half 1 -> k {8..15, 24..31}: two 16B loads at +0 / +32B.
    const __bf16* Brow = PNt + ((size_t)i * HWSZ + l16) * CCH + half * 8;

    union Frag { v16bf v; uint4 q[2]; };

    for (int kc = 0; kc < CCH; kc += KSTEP) {
        Frag af;
        {
            const uint4* ap = (const uint4*)(Arow + kc);
            af.q[0] = ap[0];            // k = 8*half + 0..7
            af.q[1] = ap[2];            // k = 8*half + 16..23
        }
        #pragma unroll
        for (int t = 0; t < NT; ++t) {
            Frag bf_;
            const uint4* bp = (const uint4*)(Brow + (size_t)t * 16 * CCH + kc);
            bf_.q[0] = bp[0];
            bf_.q[1] = bp[2];
            acc[t] = __builtin_amdgcn_wmma_f32_16x16x32_bf16(
                false, af.v, false, bf_.v, (short)0, acc[t], false, false);
        }
    }

    // Softmax over b (cols) per row a, keeping only column-sums s[b].
    // Row a = (vgpr r, lane-half): lives in 16 lanes x 9 tiles.
    float sc[NT];
    #pragma unroll
    for (int t = 0; t < NT; ++t) sc[t] = 0.f;
    #pragma unroll
    for (int r = 0; r < 8; ++r) {
        float m = acc[0][r];
        #pragma unroll
        for (int t = 1; t < NT; ++t) m = fmaxf(m, acc[t][r]);
        #pragma unroll
        for (int off = 1; off < 16; off <<= 1) m = fmaxf(m, __shfl_xor(m, off, 16));
        float e[NT], rs = 0.f;
        #pragma unroll
        for (int t = 0; t < NT; ++t) { e[t] = __expf(acc[t][r] - m); rs += e[t]; }
        #pragma unroll
        for (int off = 1; off < 16; off <<= 1) rs += __shfl_xor(rs, off, 16);
        const float inv = 1.f / rs;
        #pragma unroll
        for (int t = 0; t < NT; ++t) sc[t] += e[t] * inv;   // accumulate over rows a
    }
    #pragma unroll
    for (int t = 0; t < NT; ++t) sc[t] += __shfl_xor(sc[t], 16, 32);  // merge halves
    if (half == 0) {
        #pragma unroll
        for (int t = 0; t < NT; ++t) ldsSp[wave][t * 16 + l16] = sc[t];
    }
    __syncthreads();
    if (tid < HWSZ) {
        float s = 0.f;
        #pragma unroll
        for (int w = 0; w < NT; ++w) s += ldsSp[w][tid];    // fixed order: deterministic
        ldsS[tid] = s;
    }
    __syncthreads();

    // Fused epilogue: warped_mean -> diff^2 -> BN(eval) -> 2-class linear.
    float p0 = 0.f, p1 = 0.f;
    for (int c = tid; c < CCH; c += PAIR_THREADS) {
        const float* prow = probe + ((size_t)i * CCH + c) * HWSZ;
        float wm = 0.f;
        #pragma unroll 8
        for (int b = 0; b < HWSZ; ++b) wm = fmaf(prow[b], ldsS[b], wm);
        wm *= (1.0f / HWSZ);
        float d = wm - GM[j * CCH + c];
        d *= d;
        const float inv = bn_gamma[c] * rsqrtf(bn_var[c] + 1e-5f);
        const float bnv = (d - bn_mean[c]) * inv + bn_beta[c];
        p0 = fmaf(bnv, cls_w[c], p0);
        p1 = fmaf(bnv, cls_w[CCH + c], p1);
    }
    red0[tid] = p0; red1[tid] = p1;
    __syncthreads();
    if (tid == 0) {
        float s0 = cls_b[0], s1 = cls_b[1];
        for (int t = 0; t < PAIR_THREADS; ++t) { s0 += red0[t]; s1 += red1[t]; }
        out[pair * 2 + 0] = s0;
        out[pair * 2 + 1] = s1;
    }
}

// ---------------------------------------------------------------------------
extern "C" void kernel_launch(void* const* d_in, const int* in_sizes, int n_in,
                              void* d_out, int out_size, void* d_ws, size_t ws_size,
                              hipStream_t stream) {
    const float* probe    = (const float*)d_in[0];
    const float* gallery  = (const float*)d_in[1];
    const float* bn_gamma = (const float*)d_in[2];
    const float* bn_beta  = (const float*)d_in[3];
    const float* bn_mean  = (const float*)d_in[4];
    const float* bn_var   = (const float*)d_in[5];
    const float* cls_w    = (const float*)d_in[6];
    const float* cls_b    = (const float*)d_in[7];
    float* out = (float*)d_out;

    // Workspace (~24 MB): PNt bf16 | GNt bf16 | GM f32 | NI f32
    __bf16* PNt = (__bf16*)d_ws;                          // 8*144*2048
    __bf16* GNt = PNt + (size_t)NP * HWSZ * CCH;          // 32*144*2048
    float*  GM  = (float*)(GNt + (size_t)NG * HWSZ * CCH);// 32*2048
    float*  NIp = GM + (size_t)NG * CCH;                  // 8*144
    float*  NIg = NIp + NP * HWSZ;                        // 32*144

    norminv_kernel<<<NP, 160, 0, stream>>>(probe, NIp);
    norminv_kernel<<<NG, 160, 0, stream>>>(gallery, NIg);

    dim3 tgp(NP, HWSZ / TB, CCH / TC);
    transpose_norm_kernel<<<tgp, 256, 0, stream>>>(probe, NIp, PNt);
    dim3 tgg(NG, HWSZ / TB, CCH / TC);
    transpose_norm_kernel<<<tgg, 256, 0, stream>>>(gallery, NIg, GNt);

    gmean_kernel<<<(NG * CCH + 255) / 256, 256, 0, stream>>>(gallery, GM);

    rw_pair_kernel<<<NP * NG, PAIR_THREADS, 0, stream>>>(
        probe, PNt, GNt, GM, bn_gamma, bn_beta, bn_mean, bn_var, cls_w, cls_b, out);
}
